// GATEncoder_61280593379511
// MI455X (gfx1250) — matrix operand order repaired and checked by reading, not compiled, not run on hardware
//
#include <hip/hip_runtime.h>
#include <hip/hip_bf16.h>
#include <cstdint>

typedef __attribute__((ext_vector_type(16))) _Float16 v16h;
typedef __attribute__((ext_vector_type(8)))  _Float16 v8h;
typedef __attribute__((ext_vector_type(8)))  float    v8f;

#define HEADS 4
#define CH    64      // per-head channels (both layers)
#define FDIM  256     // HEADS*CH
#define NEG_SLOPE 0.2f
#define MT    4       // M-tiles per wave in the WMMA GEMM

__device__ __forceinline__ float lrelu(float x) { return x > 0.f ? x : NEG_SLOPE * x; }
__device__ __forceinline__ float eluf (float x) { return x > 0.f ? x : __expf(x) - 1.f; }

// float atomic max via monotone int reinterpretation (global_atomic_max_i32 / min_u32)
__device__ __forceinline__ void atomicMaxF(float* addr, float v) {
  if (v >= 0.f) atomicMax((int*)addr, __float_as_int(v));
  else          atomicMin((unsigned int*)addr, __float_as_uint(v));
}

__global__ void k_cvt_f16(const float* __restrict__ src, _Float16* __restrict__ dst, int n) {
  int i = blockIdx.x * blockDim.x + threadIdx.x;
  if (i < n) dst[i] = (_Float16)src[i];
}

// Pack W[K x Nc] (f32 row-major) into WMMA B-fragment order (f16):
//   Bp[(((ks*ntile + tn)*32 + lane)*16 + i)] = W[ks*32 + 16*(lane>>4) + i, tn*16 + (lane&15)]
// so each lane's 16 B halves for one k-step are 32 contiguous bytes.
__global__ void k_packB(const float* __restrict__ W, _Float16* __restrict__ Bp,
                        int K, int Nc) {
  int idx = blockIdx.x * blockDim.x + threadIdx.x;
  if (idx >= K * Nc) return;
  int i    = idx & 15;
  int lane = (idx >> 4) & 31;
  int tnks = idx >> 9;
  int ntile = Nc >> 4;
  int tn = tnks % ntile;
  int ks = tnks / ntile;
  int col = lane & 15, hi = lane >> 4;
  int k = ks * 32 + 16 * hi + i;
  int n = tn * 16 + col;
  Bp[idx] = (_Float16)W[(size_t)k * Nc + n];
}

// C[M x Nc] (f32) = A[M x K] (f16 row-major) * B (pre-packed fragments).
// One wave computes MT vertical 16x16 tiles of one 16-wide N strip; K fully
// unrolled (template). Tail M-tiles are clamped (duplicate benign writes) so
// there is NO divergent control flow: EXEC stays all-ones around every WMMA.
// A/B use contiguous k = 16*(lane/16) + s, not the ISA table order; C/D layout per CDNA5 ISA 7.12.2.
template <int K>
__global__ __launch_bounds__(256) void k_gemm_wmma(const _Float16* __restrict__ A,
                                                   const _Float16* __restrict__ Bp,
                                                   float* __restrict__ C,
                                                   int M, int Nc) {
  const int wave  = (int)((blockIdx.x * blockDim.x + threadIdx.x) >> 5);
  const int lane  = threadIdx.x & 31;
  const int ntile = Nc >> 4;
  const int mtiles = M >> 4;                  // M divisible by 16 here
  const int tn  = wave % ntile;
  const int tm0 = (wave / ntile) * MT;
  const int col = lane & 15;
  const int hi  = lane >> 4;

  int tmt[MT];
#pragma unroll
  for (int t = 0; t < MT; ++t) tmt[t] = min(tm0 + t, mtiles - 1);

  const _Float16* Ap[MT];
#pragma unroll
  for (int t = 0; t < MT; ++t)
    Ap[t] = A + (size_t)(tmt[t] * 16 + col) * K + 16 * hi;

  v8f acc[MT] = {};
  const _Float16* bp = Bp + ((size_t)tn * 32 + lane) * 16;
  const size_t bstride = (size_t)ntile * 32 * 16;   // halves per k-step

#pragma unroll
  for (int ks = 0; ks < K / 32; ++ks) {
    const v16h b = *(const v16h*)(bp + ks * bstride);
#pragma unroll
    for (int t = 0; t < MT; ++t) {
      const v8h alo = *(const v8h*)(Ap[t] + ks * 32);        // 16B aligned
      const v8h ahi = *(const v8h*)(Ap[t] + ks * 32 + 8);
      v16h a;
#pragma unroll
      for (int i = 0; i < 8; ++i) { a[i] = alo[i]; a[i + 8] = ahi[i]; }
      acc[t] = __builtin_amdgcn_wmma_f32_16x16x32_f16(false, a, false, b,
                                                      (short)0, acc[t], false, false);
    }
  }

#pragma unroll
  for (int t = 0; t < MT; ++t) {
    float* Cp = C + (size_t)(tmt[t] * 16 + 8 * hi) * Nc + tn * 16 + col;
#pragma unroll
    for (int r = 0; r < 8; ++r) Cp[(size_t)r * Nc] = acc[t][r];
  }
}

// a_s[n,h] = sum_c H[n,h*64+c]*att_s[h*64+c] ; same for a_d
__global__ void k_scores(const float* __restrict__ Hf,
                         const float* __restrict__ att_s, const float* __restrict__ att_d,
                         float* __restrict__ a_s, float* __restrict__ a_d, int N) {
  int i = blockIdx.x * blockDim.x + threadIdx.x;      // n*HEADS + h
  if (i >= N * HEADS) return;
  int h = i & (HEADS - 1);
  int n = i >> 2;
  const float* hp = Hf + (size_t)n * FDIM + h * CH;
  const float* as = att_s + h * CH;
  const float* ad = att_d + h * CH;
  float s = 0.f, d = 0.f;
#pragma unroll 8
  for (int c = 0; c < CH; ++c) { float v = hp[c]; s += v * as[c]; d += v * ad[c]; }
  a_s[i] = s; a_d[i] = d;
}

__global__ void k_init(float* __restrict__ acc, float* __restrict__ m,
                       float* __restrict__ z, int accN, int nmz) {
  int i = blockIdx.x * blockDim.x + threadIdx.x;
  if (i < accN) acc[i] = 0.f;
  if (i < nmz) { m[i] = -__builtin_inff(); z[i] = 0.f; }
}

__device__ __forceinline__ void edge_nodes(const int* __restrict__ srcs,
                                           const int* __restrict__ dsts,
                                           int e, int E, int& s, int& d) {
  if (e < E) { s = srcs[e]; d = dsts[e]; } else { s = e - E; d = e - E; }  // self-loops
}

__global__ void k_edge_max(const int* __restrict__ srcs, const int* __restrict__ dsts,
                           const float* __restrict__ a_s, const float* __restrict__ a_d,
                           float* __restrict__ m, int E, int EE) {
  int i = blockIdx.x * blockDim.x + threadIdx.x;
  if (i >= EE * HEADS) return;
  int e = i >> 2, h = i & 3, s, d;
  edge_nodes(srcs, dsts, e, E, s, d);
  atomicMaxF(&m[d * HEADS + h], lrelu(a_s[s * HEADS + h] + a_d[d * HEADS + h]));
}

__global__ void k_edge_sum(const int* __restrict__ srcs, const int* __restrict__ dsts,
                           const float* __restrict__ a_s, const float* __restrict__ a_d,
                           const float* __restrict__ m, float* __restrict__ z,
                           int E, int EE) {
  int i = blockIdx.x * blockDim.x + threadIdx.x;
  if (i >= EE * HEADS) return;
  int e = i >> 2, h = i & 3, s, d;
  edge_nodes(srcs, dsts, e, E, s, d);
  float v = lrelu(a_s[s * HEADS + h] + a_d[d * HEADS + h]);
  atomicAdd(&z[d * HEADS + h], __expf(v - m[d * HEADS + h]));
}

__global__ void k_edge_alpha(const int* __restrict__ srcs, const int* __restrict__ dsts,
                             const float* __restrict__ a_s, const float* __restrict__ a_d,
                             const float* __restrict__ m, const float* __restrict__ z,
                             float* __restrict__ alpha, int E, int EE) {
  int i = blockIdx.x * blockDim.x + threadIdx.x;
  if (i >= EE * HEADS) return;
  int e = i >> 2, h = i & 3, s, d;
  edge_nodes(srcs, dsts, e, E, s, d);
  float v = lrelu(a_s[s * HEADS + h] + a_d[d * HEADS + h]);
  alpha[i] = __expf(v - m[d * HEADS + h]) / (z[d * HEADS + h] + 1e-16f);
}

// 64 threads per edge (one per channel), 4 heads per thread: coalesced gather of
// H[src], L2-resident atomic scatter into acc[dst].
__global__ void k_edge_agg(const int* __restrict__ srcs, const int* __restrict__ dsts,
                           const float* __restrict__ alpha, const float* __restrict__ Hf,
                           float* __restrict__ acc, int E, int EE) {
  int i = blockIdx.x * blockDim.x + threadIdx.x;
  if (i >= EE * CH) return;
  int e = i >> 6, c = i & 63, s, d;
  edge_nodes(srcs, dsts, e, E, s, d);
  const float4 al = *(const float4*)(alpha + (size_t)e * HEADS);   // 16B aligned
  const float* hs = Hf + (size_t)s * FDIM + c;
  float* ap = acc + (size_t)d * FDIM + c;
  atomicAdd(ap,           al.x * hs[0]);
  atomicAdd(ap + 1 * CH,  al.y * hs[1 * CH]);
  atomicAdd(ap + 2 * CH,  al.z * hs[2 * CH]);
  atomicAdd(ap + 3 * CH,  al.w * hs[3 * CH]);
}

__global__ void k_epi1(const float* __restrict__ acc, const float* __restrict__ bias,
                       _Float16* __restrict__ out16, int total) {
  int i = blockIdx.x * blockDim.x + threadIdx.x;
  if (i >= total) return;
  out16[i] = (_Float16)eluf(acc[i] + bias[i & (FDIM - 1)]);
}

__global__ void k_epi2(float* __restrict__ out, const float* __restrict__ bias, int total) {
  int i = blockIdx.x * blockDim.x + threadIdx.x;
  if (i >= total) return;
  out[i] = eluf(out[i] + bias[i & (FDIM - 1)]);
}

extern "C" void kernel_launch(void* const* d_in, const int* in_sizes, int n_in,
                              void* d_out, int out_size, void* d_ws, size_t ws_size,
                              hipStream_t stream) {
  const float* x    = (const float*)d_in[0];
  const int*   ei   = (const int*)  d_in[1];
  const float* W1   = (const float*)d_in[2];
  const float* as1  = (const float*)d_in[3];
  const float* ad1  = (const float*)d_in[4];
  const float* b1   = (const float*)d_in[5];
  const float* W2   = (const float*)d_in[6];
  const float* as2  = (const float*)d_in[7];
  const float* ad2  = (const float*)d_in[8];
  const float* b2   = (const float*)d_in[9];
  float* out = (float*)d_out;

  const int N  = in_sizes[0] / 128;   // 50000
  const int E  = in_sizes[1] / 2;     // 800000
  const int EE = E + N;               // + self loops
  const int INC = 128;

  // workspace carve-out
  char* ws = (char*)d_ws; size_t off = 0;
  auto alloc = [&](size_t bytes) -> void* {
    size_t o = (off + 255) & ~(size_t)255; off = o + bytes; return (void*)(ws + o);
  };
  _Float16* Xh   = (_Float16*)alloc((size_t)N * INC * 2);
  _Float16* W1p  = (_Float16*)alloc((size_t)INC * FDIM * 2);   // packed B fragments
  _Float16* W2p  = (_Float16*)alloc((size_t)FDIM * FDIM * 2);  // packed B fragments
  float*    Hf   = (float*)   alloc((size_t)N * FDIM * 4);     // GEMM out (reused L1/L2)
  _Float16* Ho16 = (_Float16*)alloc((size_t)N * FDIM * 2);     // elu(layer1) in f16
  float*    accB = (float*)   alloc((size_t)N * FDIM * 4);     // layer-1 accumulator
  float*    a_s  = (float*)   alloc((size_t)N * HEADS * 4);
  float*    a_d  = (float*)   alloc((size_t)N * HEADS * 4);
  float*    mbuf = (float*)   alloc((size_t)N * HEADS * 4);
  float*    zbuf = (float*)   alloc((size_t)N * HEADS * 4);
  float*    alp  = (float*)   alloc((size_t)EE * HEADS * 4);

  const dim3 blk(256);
  auto grid = [](int n) { return dim3((unsigned)((n + 255) / 256)); };
  const int mtiles  = (N + 15) / 16;
  const int mgroups = (mtiles + MT - 1) / MT;
  const int waves   = mgroups * (FDIM / 16);
  const dim3 ggemm((unsigned)((waves * 32 + 255) / 256));

  const int* esrc = ei;
  const int* edst = ei + E;

  // precision convert + weight pre-pack for matrix cores
  k_cvt_f16<<<grid(N * INC),     blk, 0, stream>>>(x, Xh, N * INC);
  k_packB  <<<grid(INC * FDIM),  blk, 0, stream>>>(W1, W1p, INC, FDIM);
  k_packB  <<<grid(FDIM * FDIM), blk, 0, stream>>>(W2, W2p, FDIM, FDIM);

  // ---- layer 1 ----
  k_gemm_wmma<128><<<ggemm, blk, 0, stream>>>(Xh, W1p, Hf, N, FDIM);
  k_scores<<<grid(N * HEADS), blk, 0, stream>>>(Hf, as1, ad1, a_s, a_d, N);
  k_init<<<grid(N * FDIM), blk, 0, stream>>>(accB, mbuf, zbuf, N * FDIM, N * HEADS);
  k_edge_max  <<<grid(EE * HEADS), blk, 0, stream>>>(esrc, edst, a_s, a_d, mbuf, E, EE);
  k_edge_sum  <<<grid(EE * HEADS), blk, 0, stream>>>(esrc, edst, a_s, a_d, mbuf, zbuf, E, EE);
  k_edge_alpha<<<grid(EE * HEADS), blk, 0, stream>>>(esrc, edst, a_s, a_d, mbuf, zbuf, alp, E, EE);
  k_edge_agg  <<<grid(EE * CH),    blk, 0, stream>>>(esrc, edst, alp, Hf, accB, E, EE);
  k_epi1<<<grid(N * FDIM), blk, 0, stream>>>(accB, b1, Ho16, N * FDIM);

  // ---- layer 2 (accumulate straight into d_out) ----
  k_gemm_wmma<256><<<ggemm, blk, 0, stream>>>(Ho16, W2p, Hf, N, FDIM);
  k_scores<<<grid(N * HEADS), blk, 0, stream>>>(Hf, as2, ad2, a_s, a_d, N);
  k_init<<<grid(N * FDIM), blk, 0, stream>>>(out, mbuf, zbuf, N * FDIM, N * HEADS);
  k_edge_max  <<<grid(EE * HEADS), blk, 0, stream>>>(esrc, edst, a_s, a_d, mbuf, E, EE);
  k_edge_sum  <<<grid(EE * HEADS), blk, 0, stream>>>(esrc, edst, a_s, a_d, mbuf, zbuf, E, EE);
  k_edge_alpha<<<grid(EE * HEADS), blk, 0, stream>>>(esrc, edst, a_s, a_d, mbuf, zbuf, alp, E, EE);
  k_edge_agg  <<<grid(EE * CH),    blk, 0, stream>>>(esrc, edst, alp, Hf, out, E, EE);
  k_epi2<<<grid(N * FDIM), blk, 0, stream>>>(out, b2, N * FDIM);
}
